// QuadPrior_GenomicEngine_78391743086721
// MI455X (gfx1250) — matrix operand order repaired and checked by reading, not compiled, not run on hardware
//
#include <hip/hip_runtime.h>
#include <hip/hip_bf16.h>
#include <math.h>

#define NB 8
#define E 512
#define BS 64
#define NHEAD 8
#define HD 64
#define TSEQ 2048
#define BBATCH 2
#define NROWS (BBATCH*TSEQ)
#define VOCAB 7
#define RNK 8
#define NBUCK 16
#define EPSLN 1e-5f
#define LP 72               // LDS row pitch (bf16 elements), conflict-free for b128 gathers
#define WBLK (BS*LP)        // one pre-transposed 64x64 weight block (padded), bf16 elems
#define NURD 25             // 4 layers x 6 urdhvas + lm

typedef __attribute__((ext_vector_type(16))) __bf16 v16bf;
typedef __attribute__((ext_vector_type(8)))  __bf16 v8bf;
typedef __attribute__((ext_vector_type(2)))  __bf16 v2bf;
typedef __attribute__((ext_vector_type(8)))  float  v8f;

__device__ __forceinline__ float wave_reduce_sum(float v) {
  #pragma unroll
  for (int off = 16; off > 0; off >>= 1)
    v += __shfl_xor(v, off, 32);
  return v;
}

__device__ __forceinline__ float gelu_exact(float v) {
  return 0.5f * v * (1.f + erff(v * 0.70710678118654752f));
}

// ---------------- embedding: h = tok[x] + pos ----------------
__global__ void k_embed(const int* __restrict__ x, const float* __restrict__ tok,
                        const float* __restrict__ pos, float* __restrict__ h) {
  int i = blockIdx.x * blockDim.x + threadIdx.x;
  if (i >= NROWS * E) return;
  int n = i / E, e = i % E;
  int t = n % TSEQ;
  h[i] = tok[x[n] * E + e] + pos[t * E + e];
}

// ---------------- one-time weight transpose f32 -> bf16 (pitch LP) ----------------
// Block j of urd set u:  j in [0,8)  -> Wv[j]
//                        j in [8,15) -> Wr[j-8]
//                        j in [15,22)-> Wl[j-15]
// Stored transposed: out[t*LP + s] = W[s*64 + t]  (K index s fastest, matching ldsB)
struct WTable { const float* p[NURD * 3]; };

__global__ __launch_bounds__(256)
void k_prep_weights(WTable wt, __bf16* __restrict__ out) {
  int blk = blockIdx.x;            // 0 .. NURD*22-1
  int u = blk / 22, j = blk % 22;
  const float* W;
  if (j < 8)       W = wt.p[u * 3 + 0] + (size_t)j * BS * BS;
  else if (j < 15) W = wt.p[u * 3 + 1] + (size_t)(j - 8) * BS * BS;
  else             W = wt.p[u * 3 + 2] + (size_t)(j - 15) * BS * BS;
  __bf16* o = out + ((size_t)u * 22 + j) * WBLK;
  #pragma unroll
  for (int i = 0; i < 16; ++i) {
    int idx = i * 256 + threadIdx.x;   // 0..4095
    int s = idx >> 6, t = idx & 63;
    o[t * LP + s] = (__bf16)W[s * BS + t];
  }
}

// ---------------- layernorm: one wave32 per row ----------------
__global__ __launch_bounds__(256)
void k_layernorm(const float* __restrict__ x, const float* __restrict__ g,
                 const float* __restrict__ b, float* __restrict__ y) {
  int lane = threadIdx.x & 31;
  int wave = threadIdx.x >> 5;
  int row = blockIdx.x * 8 + wave;
  if (row >= NROWS) return;
  const float* xr = x + (size_t)row * E;
  float vals[16];
  float s = 0.f;
  #pragma unroll
  for (int i = 0; i < 16; ++i) { vals[i] = xr[i * 32 + lane]; s += vals[i]; }
  s = wave_reduce_sum(s);
  float mean = s * (1.f / E);
  float var = 0.f;
  #pragma unroll
  for (int i = 0; i < 16; ++i) { float d = vals[i] - mean; var += d * d; }
  var = wave_reduce_sum(var) * (1.f / E);
  float rstd = rsqrtf(var + EPSLN);
  float* yr = y + (size_t)row * E;
  #pragma unroll
  for (int i = 0; i < 16; ++i) {
    int e = i * 32 + lane;
    yr[e] = (vals[i] - mean) * rstd * g[e] + b[e];
  }
}

// ---------------- urdhva block-banded linear via bf16 WMMA ----------------
// y[:, b*64:(b+1)*64] = sum over kb in {b-1,b,b+1}:  x_block(kb) @ W(kb->b)  + bias
// wt: pre-transposed bf16 weights (22 blocks, pitch LP). Fused epilogue:
// +residual (res != null) or exact GELU (do_gelu).
__global__ __launch_bounds__(256)
void k_urdhva(const float* __restrict__ x, const __bf16* __restrict__ wt,
              const float* __restrict__ bias, const float* __restrict__ res,
              int do_gelu, float* __restrict__ y) {
  __shared__ alignas(16) __bf16 ldsA[64 * LP];   // activations, row-major [row][k]
  __shared__ alignas(16) __bf16 ldsB[64 * LP];   // weights transposed [col][k]
  int tid  = threadIdx.x;
  int lane = tid & 31, wave = tid >> 5;
  int waveM = wave & 3;       // 16-row strip within 64-row tile
  int waveN = wave >> 2;      // 0..1 : 32-col half
  int row0 = blockIdx.x * 64;
  int b = blockIdx.y;

  v8f acc0 = {};
  v8f acc1 = {};

  int kb_lo = (b > 0) ? b - 1 : 0;
  int kb_hi = (b < NB - 1) ? b + 1 : NB - 1;

  for (int kb = kb_lo; kb <= kb_hi; ++kb) {
    int j = (kb == b) ? b : (kb == b - 1) ? (8 + b - 1) : (15 + b);
    // stage A: 64x64 f32 -> packed bf16 pairs (8 pairs/thread)
    #pragma unroll
    for (int i = 0; i < 8; ++i) {
      int idx = i * 256 + tid;        // 0..2047 pair index
      int r = idx >> 5, p = idx & 31; // row, column pair
      float2 xv = *(const float2*)(x + (size_t)(row0 + r) * E + kb * BS + 2 * p);
      v2bf pk; pk[0] = (__bf16)xv.x; pk[1] = (__bf16)xv.y;
      *(v2bf*)(ldsA + r * LP + 2 * p) = pk;
    }
    // stage B: straight copy of pre-transposed bf16 block (2304 dwords)
    {
      const unsigned* src = (const unsigned*)(wt + (size_t)j * WBLK);
      unsigned* dst = (unsigned*)ldsB;
      #pragma unroll
      for (int i = 0; i < 9; ++i) {
        int idx = i * 256 + tid;      // 0..2303
        dst[idx] = src[idx];
      }
    }
    // prefetch next K-block's activation rows into cache (global_prefetch_b8)
    if (kb < kb_hi) {
      int r = tid & 63;
      __builtin_prefetch(x + (size_t)(row0 + r) * E + (kb + 1) * BS, 0, 1);
    }
    __syncthreads();

    #pragma unroll
    for (int kc = 0; kc < 2; ++kc) {
      int kbase = kc * 32;
      // A fragment (16x32): lanes 0-15 rows M, K {0..7,16..23}; lanes 16-31 K {8..15,24..31}
      int arow = waveM * 16 + (lane & 15);
      int ah = (lane < 16) ? 0 : 8;
      v8bf alo = *(const v8bf*)(ldsA + arow * LP + kbase + ah);
      v8bf ahi = *(const v8bf*)(ldsA + arow * LP + kbase + 16 + ah);
      v16bf afrag;
      #pragma unroll
      for (int e = 0; e < 8; ++e) { afrag[e] = alo[e]; afrag[e + 8] = ahi[e]; }
      int bh = (lane < 16) ? 0 : 8;
      #pragma unroll
      for (int nt = 0; nt < 2; ++nt) {
        // B fragment (32x16): lane holds col N, K {0..7,16..23} (lanes 0-15) or {8..15,24..31} (lanes 16-31)
        int bcol = waveN * 32 + nt * 16 + (lane & 15);
        v8bf blo = *(const v8bf*)(ldsB + bcol * LP + kbase + bh);
        v8bf bhi = *(const v8bf*)(ldsB + bcol * LP + kbase + 16 + bh);
        v16bf bfrag;
        #pragma unroll
        for (int e = 0; e < 8; ++e) { bfrag[e] = blo[e]; bfrag[e + 8] = bhi[e]; }
        if (nt == 0)
          acc0 = __builtin_amdgcn_wmma_f32_16x16x32_bf16(false, afrag, false, bfrag,
                                                         (short)0, acc0, false, false);
        else
          acc1 = __builtin_amdgcn_wmma_f32_16x16x32_bf16(false, afrag, false, bfrag,
                                                         (short)0, acc1, false, false);
      }
    }
    __syncthreads();
  }

  // D/C 16x16 f32 layout: VGPR r -> M = r (lanes 0-15) or r+8 (lanes 16-31); N = lane%16
  int mbase = row0 + waveM * 16 + ((lane < 16) ? 0 : 8);
  #pragma unroll
  for (int nt = 0; nt < 2; ++nt) {
    int col = b * 64 + waveN * 32 + nt * 16 + (lane & 15);
    float bv = bias[col];
    #pragma unroll
    for (int r = 0; r < 8; ++r) {
      float a = ((nt == 0) ? acc0[r] : acc1[r]) + bv;
      size_t off = (size_t)(mbase + r) * E + col;
      if (res) a += res[off];
      if (do_gelu) a = gelu_exact(a);
      y[off] = a;
    }
  }
}

// ---------------- sparse masked attention: one wave per (b,h,t) ----------------
__global__ __launch_bounds__(256)
void k_attn(const float* __restrict__ q, const float* __restrict__ k,
            const float* __restrict__ v, float* __restrict__ out) {
  int lane = threadIdx.x & 31;
  int w = (blockIdx.x * blockDim.x + threadIdx.x) >> 5;   // BBATCH*NHEAD*TSEQ waves
  int t  = w % TSEQ;
  int hh = (w / TSEQ) % NHEAD;
  int bb = w / (TSEQ * NHEAD);
  size_t rowq = ((size_t)bb * TSEQ + t) * E + hh * HD;
  float2 qv = *(const float2*)(q + rowq + lane * 2);

  int keys[16]; int nk = 0;
  #pragma unroll
  for (int d = 0; d < 5; ++d) if (t - d >= 0) keys[nk++] = t - d;
  for (int step = 8; step < TSEQ; step <<= 1) if (t - step >= 0) keys[nk++] = t - step;

  const float scale = 0.125f;  // 1/sqrt(64)
  float m = -INFINITY, l = 0.f;
  float ax = 0.f, ay = 0.f;
  for (int i = 0; i < nk; ++i) {
    size_t rs = ((size_t)bb * TSEQ + keys[i]) * E + hh * HD;
    float2 kv = *(const float2*)(k + rs + lane * 2);
    float sc = wave_reduce_sum(qv.x * kv.x + qv.y * kv.y) * scale;
    float nm = fmaxf(m, sc);
    float esc = __expf(m - nm);     // first iter: exp(-inf)=0
    float p   = __expf(sc - nm);
    float2 vv = *(const float2*)(v + rs + lane * 2);
    l = l * esc + p;
    ax = ax * esc + p * vv.x;
    ay = ay * esc + p * vv.y;
    m = nm;
  }
  float inv = 1.f / l;
  float2 o; o.x = ax * inv; o.y = ay * inv;
  *(float2*)(out + rowq + lane * 2) = o;
}

// ---------------- head: logits = t1 @ Wf + alpha*(hn @ pA[bucket]) @ pB ----------------
__global__ __launch_bounds__(256)
void k_head(const float* __restrict__ hn, const float* __restrict__ t1,
            const float* __restrict__ Wf, const float* __restrict__ pA,
            const float* __restrict__ pB, const float* __restrict__ alpha,
            float* __restrict__ out) {
  int lane = threadIdx.x & 31, wave = threadIdx.x >> 5;
  int row = blockIdx.x * 8 + wave;
  if (row >= NROWS) return;
  int t = row % TSEQ;
  int bidx = t / (TSEQ / NBUCK);
  if (bidx > NBUCK - 1) bidx = NBUCK - 1;
  const float* pAr = pA + (size_t)bidx * E * RNK;
  float accA[RNK] = {};
  float accW[VOCAB] = {};
  for (int i = 0; i < 16; ++i) {
    int e = i * 32 + lane;
    float hv = hn[(size_t)row * E + e];
    float tv = t1[(size_t)row * E + e];
    #pragma unroll
    for (int r = 0; r < RNK; ++r) accA[r] += hv * pAr[e * RNK + r];
    #pragma unroll
    for (int vv = 0; vv < VOCAB; ++vv) accW[vv] += tv * Wf[e * VOCAB + vv];
  }
  #pragma unroll
  for (int r = 0; r < RNK; ++r) accA[r] = wave_reduce_sum(accA[r]);
  #pragma unroll
  for (int vv = 0; vv < VOCAB; ++vv) accW[vv] = wave_reduce_sum(accW[vv]);
  if (lane == 0) {
    float al = alpha[0];
    #pragma unroll
    for (int vv = 0; vv < VOCAB; ++vv) {
      float p = 0.f;
      #pragma unroll
      for (int r = 0; r < RNK; ++r) p += accA[r] * pB[r * VOCAB + vv];
      out[(size_t)row * VOCAB + vv] = accW[vv] + al * p;
    }
  }
}

// ---------------- host orchestration ----------------
extern "C" void kernel_launch(void* const* d_in, const int* in_sizes, int n_in,
                              void* d_out, int out_size, void* d_ws, size_t ws_size,
                              hipStream_t stream) {
  (void)in_sizes; (void)n_in; (void)out_size; (void)ws_size;
  int idx = 0;
  const int* x = (const int*)d_in[idx++];
  const float* tok = (const float*)d_in[idx++];
  const float* pos = (const float*)d_in[idx++];

  struct Urd { const float *Wv, *Wr, *Wl, *b; };
  auto getf = [&]() -> const float* { return (const float*)d_in[idx++]; };
  auto getUrd = [&]() -> Urd { Urd u; u.Wv = getf(); u.Wr = getf(); u.Wl = getf(); u.b = getf(); return u; };

  struct Blk { const float *ln1g, *ln1b, *ln2g, *ln2b; Urd q, k, v, o, f1, f2; } blk[4];
  for (int L = 0; L < 4; ++L) {
    blk[L].ln1g = getf(); blk[L].ln1b = getf();
    blk[L].q = getUrd(); blk[L].k = getUrd(); blk[L].v = getUrd(); blk[L].o = getUrd();
    blk[L].ln2g = getf(); blk[L].ln2b = getf();
    blk[L].f1 = getUrd(); blk[L].f2 = getUrd();
  }
  const float* lnfg = getf();
  const float* lnfb = getf();
  Urd lm = getUrd();
  const float* Wf = getf();
  const float* pA = getf();
  const float* pB = getf();
  const float* alpha = getf();

  const size_t NE = (size_t)NROWS * E;
  float* h  = (float*)d_ws;
  float* xn = h  + NE;
  float* qb = xn + NE;
  float* kb = qb + NE;
  float* vb = kb + NE;
  float* ab = vb + NE;
  __bf16* wbuf = (__bf16*)(ab + NE);     // NURD * 22 * WBLK bf16 (~5 MB)

  // Build weight pointer table: u = L*6 + {q,k,v,o,f1,f2}, u=24 -> lm
  WTable wt;
  const Urd* uset[NURD];
  for (int L = 0; L < 4; ++L) {
    uset[L * 6 + 0] = &blk[L].q;  uset[L * 6 + 1] = &blk[L].k;
    uset[L * 6 + 2] = &blk[L].v;  uset[L * 6 + 3] = &blk[L].o;
    uset[L * 6 + 4] = &blk[L].f1; uset[L * 6 + 5] = &blk[L].f2;
  }
  uset[24] = &lm;
  for (int u = 0; u < NURD; ++u) {
    wt.p[u * 3 + 0] = uset[u]->Wv;
    wt.p[u * 3 + 1] = uset[u]->Wr;
    wt.p[u * 3 + 2] = uset[u]->Wl;
  }
  auto wptr = [&](int u) -> const __bf16* { return wbuf + (size_t)u * 22 * WBLK; };

  int nElem = (int)NE;
  dim3 bElem(256), gElem((nElem + 255) / 256);
  dim3 bLN(256),  gLN(NROWS / 8);
  dim3 bU(256),   gU(NROWS / 64, NB);
  dim3 bAt(256),  gAt((BBATCH * NHEAD * TSEQ) / 8);

  k_prep_weights<<<dim3(NURD * 22), dim3(256), 0, stream>>>(wt, wbuf);
  k_embed<<<gElem, bElem, 0, stream>>>(x, tok, pos, h);
  for (int L = 0; L < 4; ++L) {
    const Blk& B = blk[L];
    int u = L * 6;
    k_layernorm<<<gLN, bLN, 0, stream>>>(h, B.ln1g, B.ln1b, xn);
    k_urdhva<<<gU, bU, 0, stream>>>(xn, wptr(u + 0), B.q.b, nullptr, 0, qb);
    k_urdhva<<<gU, bU, 0, stream>>>(xn, wptr(u + 1), B.k.b, nullptr, 0, kb);
    k_urdhva<<<gU, bU, 0, stream>>>(xn, wptr(u + 2), B.v.b, nullptr, 0, vb);
    k_attn<<<gAt, bAt, 0, stream>>>(qb, kb, vb, ab);
    // h = h + o_proj(attn_out)   (fused residual, in place)
    k_urdhva<<<gU, bU, 0, stream>>>(ab, wptr(u + 3), B.o.b, h, 0, h);
    k_layernorm<<<gLN, bLN, 0, stream>>>(h, B.ln2g, B.ln2b, xn);
    // qb = gelu(ffn1(xn))        (fused gelu)
    k_urdhva<<<gU, bU, 0, stream>>>(xn, wptr(u + 4), B.f1.b, nullptr, 1, qb);
    // h = h + ffn2(qb)           (fused residual, in place)
    k_urdhva<<<gU, bU, 0, stream>>>(qb, wptr(u + 5), B.f2.b, h, 0, h);
  }
  k_layernorm<<<gLN, bLN, 0, stream>>>(h, lnfg, lnfb, xn);
  k_urdhva<<<gU, bU, 0, stream>>>(xn, wptr(24), lm.b, nullptr, 0, qb);
  k_head<<<gLN, bLN, 0, stream>>>(xn, qb, Wf, pA, pB, alpha, (float*)d_out);
}
